// Attention_26740466385723
// MI455X (gfx1250) — matrix operand order repaired and checked by reading, not compiled, not run on hardware
//
#include <hip/hip_runtime.h>
#include <hip/hip_bf16.h>

// ---------------------------------------------------------------------------
// CDNA5 (gfx1250) attention: v_wmma_f32_16x16x32_f16 matrix math, f32 accum,
// async global->LDS staging (ASYNCcnt) with double-buffered LDS tiles,
// f16 intermediates, WMMA-based softmax row-sum accumulation.
// wave32, 256-thread (8-wave) workgroups.
// ---------------------------------------------------------------------------

typedef _Float16 f16;
typedef __attribute__((ext_vector_type(16))) _Float16 v16h;
typedef __attribute__((ext_vector_type(8)))  float    v8f;

constexpr int BATCH  = 4;
constexpr int SEQ    = 1024;
constexpr int DMODEL = 1024;
constexpr int NHEADS = 16;
constexpr int DHEAD  = 64;
constexpr int INNER  = 1024;        // NHEADS * DHEAD
constexpr float QK_SCALE = 0.125f;  // 64^-0.5

// ---- WMMA wrapper: D = A(16x32 f16) * B(32x16 f16) + C(16x16 f32) ---------
__device__ inline v8f wmma32(v16h a, v16h b, v8f c) {
  return __builtin_amdgcn_wmma_f32_16x16x32_f16(false, a, false, b, (short)0, c,
                                                false, false);
}

// ---- Async global->LDS copies (gfx1250 GLOBAL_LOAD_ASYNC_TO_LDS_*).
// GV addressing: per-lane 64-bit global address, per-lane 32-bit LDS offset.
__device__ inline void async_copy_b32(const void* lds_ptr, const void* gptr) {
  unsigned lds_off = (unsigned)(unsigned long long)lds_ptr;
  asm volatile("global_load_async_to_lds_b32 %0, %1, off"
               :: "v"(lds_off), "v"((unsigned long long)gptr) : "memory");
}
__device__ inline void async_copy_b128(const void* lds_ptr, const void* gptr) {
  unsigned lds_off = (unsigned)(unsigned long long)lds_ptr;
  asm volatile("global_load_async_to_lds_b128 %0, %1, off"
               :: "v"(lds_off), "v"((unsigned long long)gptr) : "memory");
}
__device__ inline void wait_async() {
  asm volatile("s_wait_asynccnt 0" ::: "memory");
}

// ---- A-fragment (16x32, 16-bit): lane L<16 -> row L, K {0-7,16-23};
//      lane L>=16 -> row L-16, K {8-15,24-31}. k-contiguous source, any dtype.
template <typename T>
__device__ inline v16h load_a_frag(const T* p, int ld) {
  const int lane = threadIdx.x & 31;
  const T* r = p + (lane & 15) * ld + (lane >> 4) * 8;
  v16h a;
#pragma unroll
  for (int i = 0; i < 8; ++i) a[i] = (f16)r[i];
#pragma unroll
  for (int i = 0; i < 8; ++i) a[8 + i] = (f16)r[16 + i];
  return a;
}
// ---- B-fragment (32x16): lane L -> col L&15, K {0-7,16-23} (L<16) / {8-15,24-31} (L>=16).
//      Source stored k-contiguous ("transposed"), f32, converts on load.
__device__ inline v16h load_b_frag_f32(const float* p, int ld) {
  const int lane = threadIdx.x & 31;
  const float* r = p + (lane & 15) * ld + (lane >> 4) * 8;
  v16h b;
#pragma unroll
  for (int i = 0; i < 16; ++i) b[i] = (f16)r[i < 8 ? i : 8 + i];
  return b;
}

// ===========================================================================
// Tiled GEMM: C[M,N] = A[M,K] * W[K,N] (+bias). Block 128x64, BK=32,
// 8 waves = 4Mx2N, 32x32 per wave. Double-buffered async LDS staging.
// AT: dtype of A (f16 or f32). OT: output dtype (f16 or f32).
// ===========================================================================
#define GBM 128
#define GBN 64
#define GBK 32
#define WLD (GBK + 4)   // f32 W-tile stride; (n*WLD + 4k)*4 stays 16B aligned

template <typename AT, typename OT>
__global__ __launch_bounds__(256) void wmma_gemm_kernel(
    const AT* __restrict__ A, const float* __restrict__ W,
    OT* __restrict__ C, const float* __restrict__ bias,
    int M, int N, int K) {
  constexpr int ALD  = (sizeof(AT) == 2) ? GBK + 8 : GBK + 4;  // 16B-aligned rows
  constexpr int EPB  = 16 / (int)sizeof(AT);                   // elems per b128
  constexpr int ARB  = GBK / EPB;                              // b128 ops per row
  constexpr int AOPS = GBM * ARB / 256;                        // ops per thread

  __shared__ __align__(16) AT    As[2][GBM][ALD];  // [m][k]
  __shared__ __align__(16) float Wt[2][GBN][WLD];  // [n][k] (transposed in copy)

  const int tid  = threadIdx.x;
  const int lane = tid & 31;
  const int wave = tid >> 5;
  const int mblk = blockIdx.y * GBM;
  const int nblk = blockIdx.x * GBN;
  const int wm = (wave >> 1) * 32;
  const int wn = (wave & 1) * 32;

  // Stage one k-slab: A rows as b128, W transposed as b32 scatter.
  auto stage = [&](int bb, int k0) {
#pragma unroll
    for (int i = 0; i < AOPS; ++i) {                 // static trip count
      int e = tid + i * 256;
      int m = e / ARB, kk = (e % ARB) * EPB;
      async_copy_b128(&As[bb][m][kk], &A[(size_t)(mblk + m) * K + k0 + kk]);
    }
#pragma unroll
    for (int i = 0; i < GBK * GBN / 256; ++i) {      // 8 ops/thread
      int e = tid + i * 256;
      int kk = e >> 6, n = e & 63;
      async_copy_b32(&Wt[bb][n][kk], &W[(size_t)(k0 + kk) * N + nblk + n]);
    }
  };

  v8f acc[2][2] = {};
  const int nsteps = K / GBK;
  stage(0, 0);
  for (int kt = 0; kt < nsteps; ++kt) {
    wait_async();        // this wave's copies for tile kt landed in LDS
    __syncthreads();     // everyone's copies landed; prior buffer reads done
    if (kt + 1 < nsteps) stage((kt + 1) & 1, (kt + 1) * GBK);  // overlap
    const int bb = kt & 1;
    v16h af0 = load_a_frag(&As[bb][wm][0],      ALD);
    v16h af1 = load_a_frag(&As[bb][wm + 16][0], ALD);
    v16h bf0 = load_b_frag_f32(&Wt[bb][wn][0],      WLD);
    v16h bf1 = load_b_frag_f32(&Wt[bb][wn + 16][0], WLD);
    acc[0][0] = wmma32(af0, bf0, acc[0][0]);
    acc[0][1] = wmma32(af0, bf1, acc[0][1]);
    acc[1][0] = wmma32(af1, bf0, acc[1][0]);
    acc[1][1] = wmma32(af1, bf1, acc[1][1]);
  }

  // C layout: VGPR r -> row r (lanes 0-15) / r+8 (lanes 16-31), col = lane&15.
  const int crow = (lane >> 4) * 8;
  const int ccol = lane & 15;
#pragma unroll
  for (int j = 0; j < 2; ++j) {
    const int n = nblk + wn + j * 16 + ccol;
    const float bj = bias ? bias[n] : 0.0f;   // one load per column strip
#pragma unroll
    for (int i = 0; i < 2; ++i) {
      OT* cp = &C[(size_t)(mblk + wm + i * 16 + crow) * N + n];
#pragma unroll
      for (int r = 0; r < 8; ++r) cp[(size_t)r * N] = (OT)(acc[i][j][r] + bj);
    }
  }
}

// ===========================================================================
// Rotary embedding (interleaved pairs), in place on q (f16) and k half of kv.
// ===========================================================================
__global__ void rotary_kernel(f16* __restrict__ q, float* __restrict__ kv,
                              const float* __restrict__ emb) {
  const size_t total = (size_t)BATCH * SEQ * NHEADS * (DHEAD / 2);
  size_t idx = (size_t)blockIdx.x * blockDim.x + threadIdx.x;
  if (idx >= total) return;
  const int    p  = idx & 31;
  const size_t t  = idx >> 5;
  const int    h  = t & (NHEADS - 1);
  const size_t bn = t >> 4;
  const int    n  = (int)(bn & (SEQ - 1));

  const float e = emb[n * DHEAD + 2 * p];
  const float c = cosf(e), s = sinf(e);

  f16* qp = q + bn * (size_t)INNER + h * DHEAD + 2 * p;
  float q1 = (float)qp[0], q2 = (float)qp[1];
  qp[0] = (f16)(q1 * c - q2 * s);
  qp[1] = (f16)(q2 * c + q1 * s);

  float* kp = kv + bn * (size_t)(2 * INNER) + h * DHEAD + 2 * p;
  float k1 = kp[0], k2 = kp[1];
  kp[0] = k1 * c - k2 * s;
  kp[1] = k2 * c + k1 * s;
}

// ===========================================================================
// Flash attention: grid (SEQ/128, NHEADS, BATCH), 8 waves, 16 q-rows/wave.
// q is f16 (direct A-fragments); K/V streamed in 32-key blocks via
// double-buffered async LDS staging (V transposed in the copy).
// Softmax row-sums accumulated on the matrix pipe via an all-ones B-frag.
// ===========================================================================
#define ATT_ROWS 128
#define KB 32
#define KLD (DHEAD + 4)   // Ks stride (f32)
#define VLD (KB + 4)      // Vt stride (f32)

__global__ __launch_bounds__(256) void attn_kernel(
    const f16* __restrict__ q, const float* __restrict__ kv,
    f16* __restrict__ ctx) {
  __shared__ __align__(16) float Ks[2][KB][KLD];     // [key][d]: B-frag for S
  __shared__ __align__(16) float Vt[2][DHEAD][VLD];  // [d][key]: B-frag for O
  __shared__ __align__(16) f16   Ps[8][16][KB + 8];  // per-wave P (A-frag stage)

  const int tid = threadIdx.x, lane = tid & 31, wave = tid >> 5;
  const int b = blockIdx.z, h = blockIdx.y;
  const int qrow0 = blockIdx.x * ATT_ROWS + wave * 16;

  const f16*   qbase = q  + (size_t)b * SEQ * INNER       + h * DHEAD;
  const float* kbase = kv + (size_t)b * SEQ * (2 * INNER) + h * DHEAD;
  const float* vbase = kbase + INNER;

  auto stage = [&](int bb, int key0) {
    // K rows contiguous -> b128 (2/thread); V transposed -> b32 scatter (8/thread)
#pragma unroll
    for (int i = 0; i < KB * DHEAD / 4 / 256; ++i) {
      int e = tid + i * 256;
      int kk = e >> 4, d = (e & 15) * 4;
      async_copy_b128(&Ks[bb][kk][d],
                      kbase + (size_t)(key0 + kk) * (2 * INNER) + d);
    }
#pragma unroll
    for (int i = 0; i < KB * DHEAD / 256; ++i) {
      int e = tid + i * 256;
      int kk = e >> 6, d = e & 63;
      async_copy_b32(&Vt[bb][d][kk],
                     vbase + (size_t)(key0 + kk) * (2 * INNER) + d);
    }
  };

  // Q fragments (d 0-31, 32-63), f16 source, softmax scale folded in.
  v16h qf[2];
  {
    const int row = lane & 15, kh = lane >> 4;
    const f16* qp = qbase + (size_t)(qrow0 + row) * INNER;
#pragma unroll
    for (int c = 0; c < 2; ++c)
#pragma unroll
      for (int i = 0; i < 16; ++i) {
        int k = c * 32 + kh * 8 + (i < 8 ? i : 16 + (i - 8));
        qf[c][i] = (f16)((float)qp[k] * QK_SCALE);  // *2^-3, exact in f16
      }
  }

  // All-ones B fragment: osum = P * 1 + alpha*osum gives softmax row-sums in
  // C layout (every lane of a row holds the same sum -> no lane reduction).
  v16h ones;
#pragma unroll
  for (int i = 0; i < 16; ++i) ones[i] = (f16)1.0f;

  v8f o[4] = {};
  v8f osum = {};
  float rmax[8];
#pragma unroll
  for (int r = 0; r < 8; ++r) rmax[r] = -1e30f;

  const int nblocks = SEQ / KB;
  stage(0, 0);
  for (int kb = 0; kb < nblocks; ++kb) {
    wait_async();
    __syncthreads();
    if (kb + 1 < nblocks) stage((kb + 1) & 1, (kb + 1) * KB);  // overlap
    const int bb = kb & 1;

    // S = Q K^T : 16 rows x 32 keys, K=64 in two 32-chunks.
    v8f s0 = {}, s1 = {};
    s0 = wmma32(qf[0], load_b_frag_f32(&Ks[bb][0][0],   KLD), s0);
    s0 = wmma32(qf[1], load_b_frag_f32(&Ks[bb][0][32],  KLD), s0);
    s1 = wmma32(qf[0], load_b_frag_f32(&Ks[bb][16][0],  KLD), s1);
    s1 = wmma32(qf[1], load_b_frag_f32(&Ks[bb][16][32], KLD), s1);

    // Online softmax: rows in (vgpr r, lane-half), cols across 16 lanes.
    // Only the row-max needs a lane butterfly; row-sum goes via WMMA.
#pragma unroll
    for (int r = 0; r < 8; ++r) {
      float t = fmaxf(s0[r], s1[r]);
#pragma unroll
      for (int off = 1; off < 16; off <<= 1) t = fmaxf(t, __shfl_xor(t, off, 32));
      float mnew  = fmaxf(rmax[r], t);
      float alpha = __expf(rmax[r] - mnew);
      rmax[r] = mnew;
      s0[r] = __expf(s0[r] - mnew);
      s1[r] = __expf(s1[r] - mnew);
      o[0][r] *= alpha; o[1][r] *= alpha; o[2][r] *= alpha; o[3][r] *= alpha;
      osum[r] *= alpha;
    }

    // C-layout -> A-layout for P via per-wave LDS staging.
    {
      const int crow = (lane >> 4) * 8, ccol = lane & 15;
#pragma unroll
      for (int r = 0; r < 8; ++r) {
        Ps[wave][crow + r][ccol]      = (f16)s0[r];
        Ps[wave][crow + r][16 + ccol] = (f16)s1[r];
      }
    }
    __syncthreads();

    // O += P(16x32) * V(32x64);  osum += P * ones (row-sum on matrix pipe).
    v16h pf = load_a_frag(&Ps[wave][0][0], KB + 8);
    o[0] = wmma32(pf, load_b_frag_f32(&Vt[bb][0][0],  VLD), o[0]);
    o[1] = wmma32(pf, load_b_frag_f32(&Vt[bb][16][0], VLD), o[1]);
    o[2] = wmma32(pf, load_b_frag_f32(&Vt[bb][32][0], VLD), o[2]);
    o[3] = wmma32(pf, load_b_frag_f32(&Vt[bb][48][0], VLD), o[3]);
    osum = wmma32(pf, ones, osum);
  }

  // Normalize and store ctx[b, n, h*64 + d] as f16.
  const int crow = (lane >> 4) * 8, ccol = lane & 15;
  f16* cb = ctx + (size_t)b * SEQ * INNER + h * DHEAD;
#pragma unroll
  for (int r = 0; r < 8; ++r) {
    float inv = 1.0f / osum[r];
    f16* cp = cb + (size_t)(qrow0 + crow + r) * INNER + ccol;
#pragma unroll
    for (int j = 0; j < 4; ++j) cp[j * 16] = (f16)(o[j][r] * inv);
  }
}

// ===========================================================================
// Host launcher. Inputs: x, rotary_emb, Wq, Wkv, Wo, bo (all f32).
// Scratch: kv f32 (32MB) | q f16 (8MB) | ctx f16 (8MB) in d_ws.
// ===========================================================================
extern "C" void kernel_launch(void* const* d_in, const int* in_sizes, int n_in,
                              void* d_out, int out_size, void* d_ws, size_t ws_size,
                              hipStream_t stream) {
  const float* x   = (const float*)d_in[0];
  const float* emb = (const float*)d_in[1];
  const float* Wq  = (const float*)d_in[2];
  const float* Wkv = (const float*)d_in[3];
  const float* Wo  = (const float*)d_in[4];
  const float* bo  = (const float*)d_in[5];
  float* out = (float*)d_out;

  const size_t M = (size_t)BATCH * SEQ;   // 4096
  float* kvbuf = (float*)d_ws;            // M * 2*INNER f32
  f16*   qbuf  = (f16*)(kvbuf + M * 2 * INNER);
  f16*   ctx   = qbuf + M * INNER;

  dim3 blk(256);
  // q = x @ Wq  (f16 out)
  wmma_gemm_kernel<float, f16><<<dim3(INNER / GBN, M / GBM), blk, 0, stream>>>(
      x, Wq, qbuf, nullptr, (int)M, INNER, DMODEL);
  // kv = x @ Wkv (f32 out; V transpose staging needs 4B async granularity)
  wmma_gemm_kernel<float, float><<<dim3(2 * INNER / GBN, M / GBM), blk, 0, stream>>>(
      x, Wkv, kvbuf, nullptr, (int)M, 2 * INNER, DMODEL);
  // rotary on q (f16) and k half of kv (f32)
  {
    size_t pairs = (size_t)BATCH * SEQ * NHEADS * (DHEAD / 2);
    rotary_kernel<<<dim3((unsigned)((pairs + 255) / 256)), blk, 0, stream>>>(
        qbuf, kvbuf, emb);
  }
  // attention -> ctx (f16)
  attn_kernel<<<dim3(SEQ / ATT_ROWS, NHEADS, BATCH), blk, 0, stream>>>(
      qbuf, kvbuf, ctx);
  // out = ctx @ Wo + bo  (f16 A, f32 out)
  wmma_gemm_kernel<f16, float><<<dim3(DMODEL / GBN, M / GBM), blk, 0, stream>>>(
      ctx, Wo, out, bo, (int)M, DMODEL, INNER);
}
